// MinGRUBlock_39719857553530
// MI455X (gfx1250) — matrix operand order repaired and checked by reading, not compiled, not run on hardware
//
#include <hip/hip_runtime.h>
#include <hip/hip_bf16.h>
#include <math.h>

typedef __attribute__((ext_vector_type(16))) __bf16 v16bf;
typedef __attribute__((ext_vector_type(8)))  float  v8f;
typedef __attribute__((ext_vector_type(4)))  unsigned int u32x4;
typedef __attribute__((ext_vector_type(4)))  float  f32x4;

#define B_   8
#define T_   4096
#define D_   512
#define H_   512
#define M_   (B_ * T_)        // 32768 tokens
#define NCH  64               // chunks along T
#define TC   64               // steps per chunk

union AFrag { u32x4 q[2]; v16bf v; };
union BFrag { u32x4 q[2]; v16bf v; };

__device__ __forceinline__ unsigned short f2bf(float f) {
    union { float f; unsigned int u; } c; c.f = f;
    unsigned int u = c.u;
    unsigned int r = u + 0x7FFFu + ((u >> 16) & 1u);   // round-to-nearest-even
    return (unsigned short)(r >> 16);
}

__device__ __forceinline__ float softplus_f(float x) {
    return fmaxf(x, 0.0f) + log1pf(__expf(-fabsf(x)));
}

// logaddexp with -inf safety
__device__ __forceinline__ float laddexp(float a, float b) {
    float m = fmaxf(a, b);
    if (!(m > -INFINITY)) return -INFINITY;
    return m + __logf(__expf(a - m) + __expf(b - m));
}

// ---------------------------------------------------------------------------
// Kernel 0: pack W_gh (fp32, row-major D x 2H) into bf16 B-frag layout.
// Wpack layout: [n_tile(64)][k_blk(16)][lane(32)][32 bytes] (tile stride 16384B)
// Lane L: col = L%16, k = (L/16)*8 + 0..7, then 16 + (L/16)*8 + 0..7.
// ---------------------------------------------------------------------------
__global__ void packW_kernel(const float* __restrict__ W, unsigned short* __restrict__ Wp) {
    int idx  = blockIdx.x * blockDim.x + threadIdx.x;   // 0..32767
    int lane = idx & 31;
    int kb   = (idx >> 5) & 15;
    int nt   = idx >> 9;                                // 0..63
    int col  = nt * 16 + (lane & 15);
    int k0   = kb * 32 + (lane >> 4) * 8;
    unsigned short h[16];
#pragma unroll
    for (int j = 0; j < 16; ++j) h[j] = f2bf(W[(size_t)(k0 + j + (j >= 8 ? 8 : 0)) * (2 * H_) + col]);
    u32x4 lo, hi;
#pragma unroll
    for (int j = 0; j < 4; ++j) {
        lo[j] = (unsigned int)h[2 * j]     | ((unsigned int)h[2 * j + 1] << 16);
        hi[j] = (unsigned int)h[8 + 2 * j] | ((unsigned int)h[9 + 2 * j] << 16);
    }
    char* dst = (char*)Wp + (size_t)nt * 16384 + (size_t)kb * 1024 + (size_t)lane * 32;
    *(u32x4*)dst        = lo;
    *(u32x4*)(dst + 16) = hi;
}

// ---------------------------------------------------------------------------
// Kernel 1: LayerNorm + pack xn into bf16 A-frag layout (ISA 16-bit A 16x32).
// Apack layout: [m_tile(2048)][k_blk(16)][loadIdx(2)][dst_lane(32)][16 bytes]
// ---------------------------------------------------------------------------
__global__ __launch_bounds__(256) void lnPackA_kernel(
    const float* __restrict__ x, const float* __restrict__ gamma,
    const float* __restrict__ beta, unsigned short* __restrict__ Ap) {
    int wave = threadIdx.x >> 5;
    int lane = threadIdx.x & 31;
    int row  = blockIdx.x * 8 + wave;                  // < 32768
    const float* xr = x + (size_t)row * D_ + lane * 16;

    float v[16];
#pragma unroll
    for (int c = 0; c < 4; ++c) {
        f32x4 t = *(const f32x4*)(xr + 4 * c);
#pragma unroll
        for (int j = 0; j < 4; ++j) v[4 * c + j] = t[j];
    }
    float s = 0.f, sq = 0.f;
#pragma unroll
    for (int j = 0; j < 16; ++j) { s += v[j]; sq += v[j] * v[j]; }
#pragma unroll
    for (int m = 16; m >= 1; m >>= 1) {
        s  += __shfl_xor(s,  m);
        sq += __shfl_xor(sq, m);
    }
    float mu  = s  * (1.0f / D_);
    float var = sq * (1.0f / D_) - mu * mu;
    float rs  = rsqrtf(var + 1e-10f);

    unsigned short hb[16];
#pragma unroll
    for (int c = 0; c < 4; ++c) {
        f32x4 g = *(const f32x4*)(gamma + lane * 16 + 4 * c);
        f32x4 b = *(const f32x4*)(beta  + lane * 16 + 4 * c);
#pragma unroll
        for (int j = 0; j < 4; ++j)
            hb[4 * c + j] = f2bf((v[4 * c + j] - mu) * rs * g[j] + b[j]);
    }
    u32x4 lo, hi;
#pragma unroll
    for (int j = 0; j < 4; ++j) {
        lo[j] = (unsigned int)hb[2 * j]     | ((unsigned int)hb[2 * j + 1] << 16);
        hi[j] = (unsigned int)hb[8 + 2 * j] | ((unsigned int)hb[9 + 2 * j] << 16);
    }
    int mt = row >> 4, rl = row & 15;
    int kb = lane >> 1, part = lane & 1;               // part == loadIdx
    char* base = (char*)Ap + (size_t)mt * 16384 + (size_t)kb * 1024 + (size_t)part * 512;
    *(u32x4*)(base + rl * 16)        = lo;             // k offsets 0..7 of half
    *(u32x4*)(base + (rl + 16) * 16) = hi;             // k offsets 8..15 of half
}

// ---------------------------------------------------------------------------
// Kernel 2: bf16 WMMA GEMM + gate activations, software-pipelined.
// Wave tile: 32 rows (2 M-tiles) x (32 gate + 32 h cols) (4 N-tiles).
// Rotating double-buffered fragments so loads(kb+1) overlap WMMAs(kb).
// ---------------------------------------------------------------------------
__global__ __launch_bounds__(128) void gemmAct_kernel(
    const unsigned short* __restrict__ Ap, const unsigned short* __restrict__ Wp,
    float* __restrict__ lc, float* __restrict__ lv) {
    int wave = threadIdx.x >> 5;
    int lane = threadIdx.x & 31;
    int mblk = blockIdx.x;                             // 0..255
    int np   = blockIdx.y;                             // 0..15 (32-col pair)

    int mt0 = mblk * 8 + wave * 2;                     // first of 2 m-tiles
    const char* aBase0 = (const char*)Ap + (size_t)mt0 * 16384;
    const char* aBase1 = aBase0 + 16384;
    const char* wBase0 = (const char*)Wp + (size_t)(np * 2)      * 16384; // gate lo
    const char* wBase1 = wBase0 + 16384;                                  // gate hi
    const char* wBase2 = (const char*)Wp + (size_t)(32 + np * 2) * 16384; // h lo
    const char* wBase3 = wBase2 + 16384;

    v8f acc[2][4];
#pragma unroll
    for (int i = 0; i < 2; ++i)
#pragma unroll
        for (int j = 0; j < 4; ++j) { v8f z = {}; acc[i][j] = z; }

    AFrag a[2][2];
    BFrag bb[2][4];

    const size_t la16 = (size_t)lane * 16;
    const size_t la32 = (size_t)lane * 32;

#define LOAD_A(buf, kb)                                                          \
    do {                                                                         \
        size_t ko = (size_t)(kb) * 1024 + la16;                                  \
        a[buf][0].q[0] = *(const u32x4*)(aBase0 + ko);                           \
        a[buf][0].q[1] = *(const u32x4*)(aBase0 + ko + 512);                     \
        a[buf][1].q[0] = *(const u32x4*)(aBase1 + ko);                           \
        a[buf][1].q[1] = *(const u32x4*)(aBase1 + ko + 512);                     \
    } while (0)
#define LOAD_B(buf, kb)                                                          \
    do {                                                                         \
        size_t ko = (size_t)(kb) * 1024 + la32;                                  \
        bb[buf][0].q[0] = *(const u32x4*)(wBase0 + ko);                          \
        bb[buf][0].q[1] = *(const u32x4*)(wBase0 + ko + 16);                     \
        bb[buf][1].q[0] = *(const u32x4*)(wBase1 + ko);                          \
        bb[buf][1].q[1] = *(const u32x4*)(wBase1 + ko + 16);                     \
        bb[buf][2].q[0] = *(const u32x4*)(wBase2 + ko);                          \
        bb[buf][2].q[1] = *(const u32x4*)(wBase2 + ko + 16);                     \
        bb[buf][3].q[0] = *(const u32x4*)(wBase3 + ko);                          \
        bb[buf][3].q[1] = *(const u32x4*)(wBase3 + ko + 16);                     \
    } while (0)

    LOAD_A(0, 0);
    LOAD_B(0, 0);
#pragma unroll
    for (int kb = 0; kb < 16; ++kb) {
        const int cur = kb & 1;
        const int nxt = cur ^ 1;
        if (kb < 15) {                 // issue next-step loads before this step's math
            LOAD_A(nxt, kb + 1);
            LOAD_B(nxt, kb + 1);
        }
#pragma unroll
        for (int t = 0; t < 4; ++t) {
            acc[0][t] = __builtin_amdgcn_wmma_f32_16x16x32_bf16(
                false, a[cur][0].v, false, bb[cur][t].v, (short)0, acc[0][t], false, false);
            acc[1][t] = __builtin_amdgcn_wmma_f32_16x16x32_bf16(
                false, a[cur][1].v, false, bb[cur][t].v, (short)0, acc[1][t], false, false);
        }
    }
#undef LOAD_A
#undef LOAD_B

    int half = lane >> 4;
    int n    = lane & 15;
#pragma unroll
    for (int mt = 0; mt < 2; ++mt) {
        int row0 = (mt0 + mt) * 16;
#pragma unroll
        for (int vv = 0; vv < 8; ++vv) {
            size_t rowOff = (size_t)(row0 + vv + half * 8) * H_;
            {   // cols np*32 + n : gate acc[mt][0], h acc[mt][2]
                float kg = acc[mt][0][vv], hp = acc[mt][2][vv];
                float lcv = -softplus_f(kg);
                float lzv = -softplus_f(-kg);
                float lgv = (hp >= 0.0f) ? __logf(hp + 0.5f) : -softplus_f(-hp);
                size_t o = rowOff + np * 32 + n;
                lc[o] = lcv;
                lv[o] = lzv + lgv;
            }
            {   // cols np*32 + 16 + n : gate acc[mt][1], h acc[mt][3]
                float kg = acc[mt][1][vv], hp = acc[mt][3][vv];
                float lcv = -softplus_f(kg);
                float lzv = -softplus_f(-kg);
                float lgv = (hp >= 0.0f) ? __logf(hp + 0.5f) : -softplus_f(-hp);
                size_t o = rowOff + np * 32 + 16 + n;
                lc[o] = lcv;
                lv[o] = lzv + lgv;
            }
        }
    }
}

// ---------------------------------------------------------------------------
// Kernel 3: per-(b, chunk, h) aggregate of TC steps in log space.
// ---------------------------------------------------------------------------
__global__ __launch_bounds__(256) void aggChunk_kernel(
    const float* __restrict__ lc, const float* __restrict__ lv,
    float* __restrict__ aggLc, float* __restrict__ aggLv) {
    int b  = blockIdx.x;
    int ch = blockIdx.y;
    int h  = blockIdx.z * 256 + threadIdx.x;
    size_t base = ((size_t)b * T_ + (size_t)ch * TC) * H_ + h;
    float sLc = 0.0f, sLv = -INFINITY;
#pragma unroll 4
    for (int t = 0; t < TC; ++t) {
        float c = lc[base + (size_t)t * H_];
        float w = lv[base + (size_t)t * H_];
        sLv = laddexp(sLv + c, w);
        sLc += c;
    }
    size_t o = ((size_t)b * NCH + ch) * H_ + h;
    aggLc[o] = sLc;
    aggLv[o] = sLv;
}

// ---------------------------------------------------------------------------
// Kernel 4: exclusive scan of chunk aggregates per channel (b, h).
// ---------------------------------------------------------------------------
__global__ __launch_bounds__(256) void scanChunks_kernel(
    const float* __restrict__ aggLc, const float* __restrict__ aggLv,
    float* __restrict__ prefix) {
    int idx = blockIdx.x * 256 + threadIdx.x;          // 0..4095 = b*H + h
    int b = idx >> 9;
    int h = idx & 511;
    float p = -INFINITY;
    for (int ch = 0; ch < NCH; ++ch) {
        size_t o = ((size_t)b * NCH + ch) * H_ + h;
        prefix[o] = p;
        p = laddexp(p + aggLc[o], aggLv[o]);
    }
}

// ---------------------------------------------------------------------------
// Kernel 5: final pass — re-walk chunk with its prefix, emit out = h + x and
// hidden = h[:, -1].
// ---------------------------------------------------------------------------
__global__ __launch_bounds__(256) void finalScan_kernel(
    const float* __restrict__ lc, const float* __restrict__ lv,
    const float* __restrict__ prefix, const float* __restrict__ x,
    float* __restrict__ out, float* __restrict__ hidden) {
    int b  = blockIdx.x;
    int ch = blockIdx.y;
    int h  = blockIdx.z * 256 + threadIdx.x;
    float lh = prefix[((size_t)b * NCH + ch) * H_ + h];
    size_t base = ((size_t)b * T_ + (size_t)ch * TC) * H_ + h;
#pragma unroll 4
    for (int t = 0; t < TC; ++t) {
        size_t o = base + (size_t)t * H_;
        lh = laddexp(lh + lc[o], lv[o]);
        out[o] = __expf(lh) + x[o];
    }
    if (ch == NCH - 1) hidden[(size_t)b * H_ + h] = __expf(lh);
}

// ---------------------------------------------------------------------------
extern "C" void kernel_launch(void* const* d_in, const int* in_sizes, int n_in,
                              void* d_out, int out_size, void* d_ws, size_t ws_size,
                              hipStream_t stream) {
    const float* x     = (const float*)d_in[0];   // (8,4096,512)
    const float* gamma = (const float*)d_in[1];   // (512,)
    const float* beta  = (const float*)d_in[2];   // (512,)
    const float* W     = (const float*)d_in[3];   // (512,1024)

    char* ws = (char*)d_ws;
    size_t offW    = 0;                                  // Wpack bf16: 1 MB
    size_t offA    = offW + (size_t)1  * 1024 * 1024;    // Apack bf16: 32 MB
    size_t offLc   = offA + (size_t)32 * 1024 * 1024;    // lc f32:     64 MB
    size_t offLv   = offLc + (size_t)64 * 1024 * 1024;   // lv f32:     64 MB
    size_t offALc  = offLv + (size_t)64 * 1024 * 1024;   // aggLc:       1 MB
    size_t offALv  = offALc + (size_t)1 * 1024 * 1024;   // aggLv:       1 MB
    size_t offPref = offALv + (size_t)1 * 1024 * 1024;   // prefix:      1 MB

    unsigned short* Wp   = (unsigned short*)(ws + offW);
    unsigned short* Ap   = (unsigned short*)(ws + offA);
    float*          lc   = (float*)(ws + offLc);
    float*          lv   = (float*)(ws + offLv);
    float*          aLc  = (float*)(ws + offALc);
    float*          aLv  = (float*)(ws + offALv);
    float*          pref = (float*)(ws + offPref);

    float* out    = (float*)d_out;
    float* hidden = out + (size_t)B_ * T_ * D_;

    packW_kernel<<<128, 256, 0, stream>>>(W, Wp);
    lnPackA_kernel<<<M_ / 8, 256, 0, stream>>>(x, gamma, beta, Ap);
    gemmAct_kernel<<<dim3(M_ / 128, 16), 128, 0, stream>>>(Ap, Wp, lc, lv);
    aggChunk_kernel<<<dim3(B_, NCH, 2), 256, 0, stream>>>(lc, lv, aLc, aLv);
    scanChunks_kernel<<<(B_ * H_) / 256, 256, 0, stream>>>(aLc, aLv, pref);
    finalScan_kernel<<<dim3(B_, NCH, 2), 256, 0, stream>>>(lc, lv, pref, x, out, hidden);
}
